// GraphAgent_28896539967835
// MI455X (gfx1250) — matrix operand order repaired and checked by reading, not compiled, not run on hardware
//
#include <hip/hip_runtime.h>

typedef unsigned short u16;
typedef unsigned int u32;
typedef unsigned long long u64;
typedef __attribute__((ext_vector_type(16))) __bf16 v16bf;
typedef __attribute__((ext_vector_type(8)))  float  v8f;

#define NEMB 128
#define NG   1024        // graphs
#define NPG  16
#define NNODE (NG*NPG)   // 16384
#define NEDGE (NG*16)    // 16384
#define NSTEM (NG*4)     // 4096
#define NSTEPS 12
#define OPS  105         // out per stem

union FragU { uint4 u[2]; v16bf v; };

__device__ __forceinline__ u16 f2bf(float f) {
  u32 u = __float_as_uint(f);
  u32 r = u + 0x7FFFu + ((u >> 16) & 1u);
  return (u16)(r >> 16);
}
__device__ __forceinline__ float lrelu(float x) { return x > 0.f ? x : 0.01f * x; }
__device__ __forceinline__ float sigm(float x) { return 1.f / (1.f + __expf(-x)); }

// A fragment: 16x32 bf16 tile, rows r0..r0+15, K = kt*32..kt*32+31, from
// row-major bf16 storage with rowStride elements.
// lane<16: row=lane,   K chunks [0..7],[16..23]; lane>=16: row=lane-16, K [8..15],[24..31]
__device__ __forceinline__ v16bf load_a_frag(const u16* base, int rowStride,
                                             int r0, int kt, int lane) {
  int r = r0 + (lane & 15);
  int half = lane >> 4;
  const u16* p = base + r * rowStride + kt * 32 + half * 8;
  FragU f;
  f.u[0] = *reinterpret_cast<const uint4*>(p);
  f.u[1] = *reinterpret_cast<const uint4*>(p + 16);
  return f.v;
}

// B fragment from pre-packed layout: [kt][ct][lane][16 bf16] contiguous.
__device__ __forceinline__ v16bf load_b_frag(const u16* packed, int nTiles,
                                             int kt, int ct, int lane) {
  const u16* p = packed + ((((kt * nTiles + ct) * 32) + lane) << 4);
  FragU f;
  f.u[0] = *reinterpret_cast<const uint4*>(p);
  f.u[1] = *reinterpret_cast<const uint4*>(p + 8);
  return f.v;
}

__device__ __forceinline__ v8f wmma_bf(v16bf a, v16bf b, v8f c) {
  return __builtin_amdgcn_wmma_f32_16x16x32_bf16(false, a, false, b, (short)0, c,
                                                 false, false);
}

// Async bulk copy of one 16-byte chunk global -> LDS (tracked by ASYNCcnt).
// ldsOff is the LDS-relative byte offset (low 32 bits of the generic pointer).
__device__ __forceinline__ void async_g2l_b128(u32 ldsOff, const void* gptr) {
  asm volatile("global_load_async_to_lds_b128 %0, %1, off"
               :: "v"(ldsOff), "v"((u64)(size_t)gptr)
               : "memory");
}
__device__ __forceinline__ void wait_async0() {
  asm volatile("s_wait_asynccnt 0" ::: "memory");
}

// ---- pack f32 weight into WMMA B-fragment bf16 layout ---------------------
// logical B[k][n]: transpose ? W[n*K+k] : W[k*Nreal+n];   n >= Nreal -> 0
__global__ void pack_b(const float* __restrict__ W, u16* __restrict__ out,
                       int K, int Nreal, int nTiles, int transpose) {
  int tid = blockIdx.x * blockDim.x + threadIdx.x;
  int total = (K >> 5) * nTiles * 512;
  if (tid >= total) return;
  int i = tid & 15;
  int lane = (tid >> 4) & 31;
  int ct = (tid >> 9) % nTiles;
  int kt = tid / (512 * nTiles);
  int k = kt * 32 + (i >> 3) * 16 + (lane >> 4) * 8 + (i & 7);
  int n = ct * 16 + (lane & 15);
  float v = 0.f;
  if (n < Nreal) v = transpose ? W[n * K + k] : W[k * Nreal + n];
  out[tid] = f2bf(v);
}

// ---- reciprocal clipped degree (edges are grouped per graph) --------------
__global__ void k_rdeg(const int* __restrict__ edge_dst, float* __restrict__ rdeg) {
  int g = blockIdx.x;
  int l = threadIdx.x;
  if (l < NPG) {
    int node = g * NPG + l;
    int c = 0;
    for (int e = 0; e < 16; ++e) c += (edge_dst[g * 16 + e] == node);
    rdeg[node] = 1.0f / (float)(c < 1 ? 1 : c);
  }
}

// ---- prologue: x = emb_block[x_ids]; state = b2e MLP(x) -------------------
__global__ void k_prologue(const float* __restrict__ emb_block,
                           const int* __restrict__ x_ids,
                           const u16* __restrict__ w1p, const float* __restrict__ b1,
                           const u16* __restrict__ w2p, const float* __restrict__ b2,
                           float* __restrict__ state_f32, u16* __restrict__ state_bf) {
  __shared__ alignas(16) u16 x_lds[16 * NEMB];
  __shared__ alignas(16) u16 t1_lds[16 * NEMB];
  int r0 = blockIdx.x * 16;
  int t = threadIdx.x, w = t >> 5, lane = t & 31;
  int half = lane >> 4, nc = lane & 15;
  for (int idx = t; idx < 16 * NEMB; idx += 128) {
    int row = idx >> 7, c = idx & 127;
    x_lds[idx] = f2bf(emb_block[x_ids[r0 + row] * NEMB + c]);
  }
  __syncthreads();
  for (int cti = 0; cti < 2; ++cti) {
    int ct = w * 2 + cti;
    v8f acc = {0.f, 0.f, 0.f, 0.f, 0.f, 0.f, 0.f, 0.f};
    for (int kt = 0; kt < 4; ++kt)
      acc = wmma_bf(load_a_frag(x_lds, NEMB, 0, kt, lane),
                    load_b_frag(w1p, 8, kt, ct, lane), acc);
    int col = ct * 16 + nc;
    for (int v = 0; v < 8; ++v)
      t1_lds[(v + 8 * half) * NEMB + col] = f2bf(lrelu(acc[v] + b1[col]));
  }
  __syncthreads();
  for (int cti = 0; cti < 2; ++cti) {
    int ct = w * 2 + cti;
    v8f acc = {0.f, 0.f, 0.f, 0.f, 0.f, 0.f, 0.f, 0.f};
    for (int kt = 0; kt < 4; ++kt)
      acc = wmma_bf(load_a_frag(t1_lds, NEMB, 0, kt, lane),
                    load_b_frag(w2p, 8, kt, ct, lane), acc);
    int col = ct * 16 + nc;
    for (int v = 0; v < 8; ++v) {
      int g = (r0 + v + 8 * half) * NEMB + col;
      float val = acc[v] + b2[col];
      state_f32[g] = val;
      state_bf[g] = f2bf(val);
    }
  }
}

// ---- per-step edge message + per-graph scatter-mean -----------------------
// msg_e = dot(state[src_e], emb_bond[b0_e]) * emb_bond[b1_e]; agg = mean@dst
__global__ void k_edge(const float* __restrict__ state_f32,
                       const float* __restrict__ emb_bond,
                       const int* __restrict__ bond_ids,
                       const int* __restrict__ edge_src,
                       const int* __restrict__ edge_dst,
                       const float* __restrict__ rdeg, float* __restrict__ agg) {
  __shared__ float s_part[16][8];
  __shared__ float s_e[16];
  __shared__ int dstloc[16];
  __shared__ int b1row[16];
  __shared__ float agg_l[16 * NEMB];
  int g = blockIdx.x, t = threadIdx.x;
  int e = t >> 3, part = t & 7;
  {
    int ge = g * 16 + e;
    int src = edge_src[ge];
    int b0 = bond_ids[ge * 2 + 0];
    const float* xr = state_f32 + src * NEMB + part * 16;
    const float* br = emb_bond + b0 * NEMB + part * 16;
    float acc = 0.f;
    for (int j = 0; j < 16; ++j) acc += xr[j] * br[j];
    s_part[e][part] = acc;
    if (part == 0) {
      dstloc[e] = edge_dst[ge] - g * NPG;
      b1row[e] = bond_ids[ge * 2 + 1];
    }
  }
  for (int idx = t; idx < 16 * NEMB; idx += 128) agg_l[idx] = 0.f;
  __syncthreads();
  if (t < 16) {
    float s = 0.f;
    for (int p = 0; p < 8; ++p) s += s_part[t][p];
    s_e[t] = s;
  }
  __syncthreads();
  int f = t;  // column owned by this thread
  for (int e2 = 0; e2 < 16; ++e2)
    agg_l[dstloc[e2] * NEMB + f] += s_e[e2] * emb_bond[b1row[e2] * NEMB + f];
  for (int v = 0; v < NPG; ++v) {
    int node = g * NPG + v;
    agg[node * NEMB + f] = agg_l[v * NEMB + f] * rdeg[node];
  }
}

// ---- per-step: m = lrelu(state@root + agg + cb); GRU(m, h=state) ----------
__global__ void k_step(float* __restrict__ state_f32, u16* __restrict__ state_bf,
                       const float* __restrict__ agg,
                       const u16* __restrict__ rootp, const float* __restrict__ cbias,
                       const u16* __restrict__ wihp, const u16* __restrict__ whhp,
                       const float* __restrict__ bih, const float* __restrict__ bhh) {
  __shared__ alignas(16) u16 m_lds[16 * NEMB];
  __shared__ alignas(16) float agg_lds[16 * NEMB];
  __shared__ float gi_lds[16 * 384];
  __shared__ float gh_lds[16 * 384];
  int r0 = blockIdx.x * 16;
  int t = threadIdx.x, w = t >> 5, lane = t & 31;
  int half = lane >> 4, nc = lane & 15;

  // Kick off async DMA of the 16x128 f32 agg tile into LDS; it overlaps the
  // A-fragment loads and the first two GEMMs below (tracked by ASYNCcnt).
  for (int i = 0; i < 4; ++i) {
    int c16 = t + i * 128;  // 512 chunks of 16B
    const float* gsrc = agg + (size_t)r0 * NEMB + c16 * 4;
    u32 loff = (u32)(size_t)(agg_lds + c16 * 4);
    async_g2l_b128(loff, gsrc);
  }

  v16bf a_s[4];
  for (int kt = 0; kt < 4; ++kt) a_s[kt] = load_a_frag(state_bf, NEMB, r0, kt, lane);

  // root GEMM: accumulate only (epilogue deferred until agg tile lands)
  v8f accm[2];
  for (int cti = 0; cti < 2; ++cti) {
    int ct = w * 2 + cti;
    v8f acc = {0.f, 0.f, 0.f, 0.f, 0.f, 0.f, 0.f, 0.f};
    for (int kt = 0; kt < 4; ++kt)
      acc = wmma_bf(a_s[kt], load_b_frag(rootp, 8, kt, ct, lane), acc);
    accm[cti] = acc;
  }
  // gh = h @ whh^T   (h == state, reuse a_s) — independent of agg
  for (int cti = 0; cti < 6; ++cti) {
    int ct = w * 6 + cti;
    v8f acc = {0.f, 0.f, 0.f, 0.f, 0.f, 0.f, 0.f, 0.f};
    for (int kt = 0; kt < 4; ++kt)
      acc = wmma_bf(a_s[kt], load_b_frag(whhp, 24, kt, ct, lane), acc);
    int col = ct * 16 + nc;
    for (int v = 0; v < 8; ++v) gh_lds[(v + 8 * half) * 384 + col] = acc[v];
  }
  wait_async0();
  __syncthreads();
  // m = lrelu(state@root + agg + bias)  -> bf16 in LDS
  for (int cti = 0; cti < 2; ++cti) {
    int ct = w * 2 + cti;
    int col = ct * 16 + nc;
    for (int v = 0; v < 8; ++v) {
      int row = v + 8 * half;
      float val = lrelu(accm[cti][v] + agg_lds[row * NEMB + col] + cbias[col]);
      m_lds[row * NEMB + col] = f2bf(val);
    }
  }
  __syncthreads();
  // gi = m @ wih^T
  v16bf a_m[4];
  for (int kt = 0; kt < 4; ++kt) a_m[kt] = load_a_frag(m_lds, NEMB, 0, kt, lane);
  for (int cti = 0; cti < 6; ++cti) {
    int ct = w * 6 + cti;
    v8f acc = {0.f, 0.f, 0.f, 0.f, 0.f, 0.f, 0.f, 0.f};
    for (int kt = 0; kt < 4; ++kt)
      acc = wmma_bf(a_m[kt], load_b_frag(wihp, 24, kt, ct, lane), acc);
    int col = ct * 16 + nc;
    for (int v = 0; v < 8; ++v) gi_lds[(v + 8 * half) * 384 + col] = acc[v];
  }
  __syncthreads();
  // GRU elementwise update (in-place on this block's 16 rows)
  for (int idx = t; idx < 16 * NEMB; idx += 128) {
    int row = idx >> 7, c = idx & 127;
    int b = row * 384;
    float r = sigm(gi_lds[b + c] + bih[c] + gh_lds[b + c] + bhh[c]);
    float z = sigm(gi_lds[b + 128 + c] + bih[128 + c] + gh_lds[b + 128 + c] + bhh[128 + c]);
    float n = tanhf(gi_lds[b + 256 + c] + bih[256 + c] +
                    r * (gh_lds[b + 256 + c] + bhh[256 + c]));
    int gidx = (r0 + row) * NEMB + c;
    float hp = state_f32[gidx];
    float hn = (1.f - z) * n + z * hp;
    state_f32[gidx] = hn;
    state_bf[gidx] = f2bf(hn);
  }
}

// ---- stem head: 3-layer MLP on concat(out[stem_idx], stem_emb) ------------
__global__ void k_stem(const float* __restrict__ state_f32,
                       const float* __restrict__ emb_stem,
                       const int* __restrict__ stemtype_ids,
                       const int* __restrict__ stems_batch,
                       const int* __restrict__ stems_local,
                       const int* __restrict__ node_offsets,
                       const u16* __restrict__ w1p, const float* __restrict__ b1,
                       const u16* __restrict__ w2p, const float* __restrict__ b2,
                       const u16* __restrict__ w3p, const float* __restrict__ b3,
                       float* __restrict__ out) {
  __shared__ alignas(16) u16 sc_lds[16 * 256];
  __shared__ alignas(16) u16 h1_lds[16 * NEMB];
  __shared__ alignas(16) u16 h2_lds[16 * NEMB];
  __shared__ int rowidx[16];
  int s0 = blockIdx.x * 16;
  int t = threadIdx.x, w = t >> 5, lane = t & 31;
  int half = lane >> 4, nc = lane & 15;
  if (t < 16) {
    int s = s0 + t;
    rowidx[t] = node_offsets[stems_batch[s]] + stems_local[s];
  }
  __syncthreads();
  for (int idx = t; idx < 16 * 256; idx += 128) {
    int row = idx >> 8, c = idx & 255;
    float v = (c < 128) ? state_f32[rowidx[row] * NEMB + c]
                        : emb_stem[stemtype_ids[s0 + row] * NEMB + (c - 128)];
    sc_lds[idx] = f2bf(v);
  }
  __syncthreads();
  for (int cti = 0; cti < 2; ++cti) {  // K=256
    int ct = w * 2 + cti;
    v8f acc = {0.f, 0.f, 0.f, 0.f, 0.f, 0.f, 0.f, 0.f};
    for (int kt = 0; kt < 8; ++kt)
      acc = wmma_bf(load_a_frag(sc_lds, 256, 0, kt, lane),
                    load_b_frag(w1p, 8, kt, ct, lane), acc);
    int col = ct * 16 + nc;
    for (int v = 0; v < 8; ++v)
      h1_lds[(v + 8 * half) * NEMB + col] = f2bf(lrelu(acc[v] + b1[col]));
  }
  __syncthreads();
  for (int cti = 0; cti < 2; ++cti) {
    int ct = w * 2 + cti;
    v8f acc = {0.f, 0.f, 0.f, 0.f, 0.f, 0.f, 0.f, 0.f};
    for (int kt = 0; kt < 4; ++kt)
      acc = wmma_bf(load_a_frag(h1_lds, NEMB, 0, kt, lane),
                    load_b_frag(w2p, 8, kt, ct, lane), acc);
    int col = ct * 16 + nc;
    for (int v = 0; v < 8; ++v)
      h2_lds[(v + 8 * half) * NEMB + col] = f2bf(lrelu(acc[v] + b2[col]));
  }
  __syncthreads();
  for (int ct = w; ct < 7; ct += 4) {  // N padded 105 -> 112
    v8f acc = {0.f, 0.f, 0.f, 0.f, 0.f, 0.f, 0.f, 0.f};
    for (int kt = 0; kt < 4; ++kt)
      acc = wmma_bf(load_a_frag(h2_lds, NEMB, 0, kt, lane),
                    load_b_frag(w3p, 7, kt, ct, lane), acc);
    int col = ct * 16 + nc;
    if (col < OPS)
      for (int v = 0; v < 8; ++v)
        out[(s0 + v + 8 * half) * OPS + col] = acc[v] + b3[col];
  }
}

// ---- global mean pool ------------------------------------------------------
__global__ void k_mean(const float* __restrict__ state_f32,
                       float* __restrict__ gmean, u16* __restrict__ gmean_bf) {
  int g = blockIdx.x, f = threadIdx.x;
  float s = 0.f;
  for (int v = 0; v < NPG; ++v) s += state_f32[(g * NPG + v) * NEMB + f];
  s *= (1.f / (float)NPG);
  gmean[g * NEMB + f] = s;
  gmean_bf[g * NEMB + f] = f2bf(s);
}

// ---- mol head layer 1 (WMMA) ----------------------------------------------
__global__ void k_g2p1(const u16* __restrict__ gmean_bf,
                       const u16* __restrict__ w1p, const float* __restrict__ b1,
                       float* __restrict__ tbuf) {
  int r0 = blockIdx.x * 16;
  int t = threadIdx.x, w = t >> 5, lane = t & 31;
  int half = lane >> 4, nc = lane & 15;
  v16bf a[4];
  for (int kt = 0; kt < 4; ++kt) a[kt] = load_a_frag(gmean_bf, NEMB, r0, kt, lane);
  for (int cti = 0; cti < 2; ++cti) {
    int ct = w * 2 + cti;
    v8f acc = {0.f, 0.f, 0.f, 0.f, 0.f, 0.f, 0.f, 0.f};
    for (int kt = 0; kt < 4; ++kt)
      acc = wmma_bf(a[kt], load_b_frag(w1p, 8, kt, ct, lane), acc);
    int col = ct * 16 + nc;
    for (int v = 0; v < 8; ++v)
      tbuf[(r0 + v + 8 * half) * NEMB + col] = lrelu(acc[v] + b1[col]);
  }
}

// ---- mol head layer 2: [G,128] @ [128,1] via wave32 reduction -------------
__global__ void k_g2p2(const float* __restrict__ tbuf, const float* __restrict__ w2,
                       const float* __restrict__ b2, float* __restrict__ out) {
  int row = blockIdx.x * 4 + (threadIdx.x >> 5);
  int lane = threadIdx.x & 31;
  float s = 0.f;
  for (int j = 0; j < 4; ++j) {
    int k = lane + 32 * j;
    s += tbuf[row * NEMB + k] * w2[k];
  }
  for (int off = 16; off >= 1; off >>= 1) s += __shfl_xor(s, off, 32);
  if (lane == 0) out[row] = s + b2[0];
}

extern "C" void kernel_launch(void* const* d_in, const int* in_sizes, int n_in,
                              void* d_out, int out_size, void* d_ws, size_t ws_size,
                              hipStream_t stream) {
  (void)in_sizes; (void)n_in; (void)out_size; (void)ws_size;
  const float* emb_block = (const float*)d_in[0];
  const float* emb_stem  = (const float*)d_in[1];
  const float* emb_bond  = (const float*)d_in[2];
  const float* conv_root = (const float*)d_in[3];
  const float* conv_bias = (const float*)d_in[4];
  const float* b2e_w1 = (const float*)d_in[5];
  const float* b2e_b1 = (const float*)d_in[6];
  const float* b2e_w2 = (const float*)d_in[7];
  const float* b2e_b2 = (const float*)d_in[8];
  const float* gru_wih = (const float*)d_in[9];
  const float* gru_whh = (const float*)d_in[10];
  const float* gru_bih = (const float*)d_in[11];
  const float* gru_bhh = (const float*)d_in[12];
  const float* s2p_w1 = (const float*)d_in[13];
  const float* s2p_b1 = (const float*)d_in[14];
  const float* s2p_w2 = (const float*)d_in[15];
  const float* s2p_b2 = (const float*)d_in[16];
  const float* s2p_w3 = (const float*)d_in[17];
  const float* s2p_b3 = (const float*)d_in[18];
  const float* g2p_w1 = (const float*)d_in[19];
  const float* g2p_b1 = (const float*)d_in[20];
  const float* g2p_w2 = (const float*)d_in[21];
  const float* g2p_b2 = (const float*)d_in[22];
  const int* x_ids = (const int*)d_in[23];
  const int* stemtype_ids = (const int*)d_in[24];
  const int* bond_ids = (const int*)d_in[25];
  const int* edge_src = (const int*)d_in[26];
  const int* edge_dst = (const int*)d_in[27];
  const int* stems_batch = (const int*)d_in[29];
  const int* stems_local = (const int*)d_in[30];
  const int* node_offsets = (const int*)d_in[31];

  char* ws = (char*)d_ws;
  size_t off = 0;
  auto carve = [&](size_t bytes) -> void* {
    void* p = ws + off;
    off += (bytes + 255) & ~(size_t)255;
    return p;
  };
  float* state_f32 = (float*)carve((size_t)NNODE * NEMB * 4);
  u16*   state_bf  = (u16*)carve((size_t)NNODE * NEMB * 2);
  float* agg       = (float*)carve((size_t)NNODE * NEMB * 4);
  float* rdeg      = (float*)carve((size_t)NNODE * 4);
  float* gmean     = (float*)carve((size_t)NG * NEMB * 4);
  u16*   gmean_bf  = (u16*)carve((size_t)NG * NEMB * 2);
  float* tbuf      = (float*)carve((size_t)NG * NEMB * 4);
  u16* rootp  = (u16*)carve(16384 * 2);
  u16* wihp   = (u16*)carve(49152 * 2);
  u16* whhp   = (u16*)carve(49152 * 2);
  u16* b2e1p  = (u16*)carve(16384 * 2);
  u16* b2e2p  = (u16*)carve(16384 * 2);
  u16* s2p1p  = (u16*)carve(32768 * 2);
  u16* s2p2p  = (u16*)carve(16384 * 2);
  u16* s2p3p  = (u16*)carve(14336 * 2);
  u16* g2p1p  = (u16*)carve(16384 * 2);

  pack_b<<<64, 256, 0, stream>>>(conv_root, rootp, 128, 128, 8, 0);
  pack_b<<<192, 256, 0, stream>>>(gru_wih, wihp, 128, 384, 24, 1);
  pack_b<<<192, 256, 0, stream>>>(gru_whh, whhp, 128, 384, 24, 1);
  pack_b<<<64, 256, 0, stream>>>(b2e_w1, b2e1p, 128, 128, 8, 0);
  pack_b<<<64, 256, 0, stream>>>(b2e_w2, b2e2p, 128, 128, 8, 0);
  pack_b<<<128, 256, 0, stream>>>(s2p_w1, s2p1p, 256, 128, 8, 0);
  pack_b<<<64, 256, 0, stream>>>(s2p_w2, s2p2p, 128, 128, 8, 0);
  pack_b<<<56, 256, 0, stream>>>(s2p_w3, s2p3p, 128, 105, 7, 0);
  pack_b<<<64, 256, 0, stream>>>(g2p_w1, g2p1p, 128, 128, 8, 0);

  k_rdeg<<<NG, 32, 0, stream>>>(edge_dst, rdeg);
  k_prologue<<<NNODE / 16, 128, 0, stream>>>(emb_block, x_ids, b2e1p, b2e_b1,
                                             b2e2p, b2e_b2, state_f32, state_bf);
  for (int s = 0; s < NSTEPS; ++s) {
    k_edge<<<NG, 128, 0, stream>>>(state_f32, emb_bond, bond_ids, edge_src,
                                   edge_dst, rdeg, agg);
    k_step<<<NNODE / 16, 128, 0, stream>>>(state_f32, state_bf, agg, rootp,
                                           conv_bias, wihp, whhp, gru_bih, gru_bhh);
  }
  k_stem<<<NSTEM / 16, 128, 0, stream>>>(state_f32, emb_stem, stemtype_ids,
                                         stems_batch, stems_local, node_offsets,
                                         s2p1p, s2p_b1, s2p2p, s2p_b2, s2p3p,
                                         s2p_b3, (float*)d_out);
  k_mean<<<NG, 128, 0, stream>>>(state_f32, gmean, gmean_bf);
  k_g2p1<<<NG / 16, 128, 0, stream>>>(gmean_bf, g2p1p, g2p_b1, tbuf);
  k_g2p2<<<NG / 4, 128, 0, stream>>>(tbuf, g2p_w2, g2p_b2,
                                     (float*)d_out + (size_t)NSTEM * OPS);
}
